// MultiResolutionSpectralLoss_66683662238478
// MI455X (gfx1250) — compile-verified, matrix operand order checked by reading, not run on hardware
//
#include <hip/hip_runtime.h>
#include <math.h>

// ---------------------------------------------------------------------------
// MI455X (gfx1250) multi-resolution log-mel spectral loss.
//  - LDS-resident radix-2 FFT, one workgroup per frame (twiddles cached in LDS)
//  - mel projection: D = W(228xK) x spec^T  as bf16 WMMA GEMM, f32 accum,
//    LDS-staged operands (conflict-free 72B row stride), coalesced uint2 loads
//  - 8 accumulator tiles per wave (64 VGPRs) -> no scratch spills
//  - workspace peak ~74 MB (fits comfortably in the 192 MB L2)
// ---------------------------------------------------------------------------

typedef __bf16 bf16_t;
typedef __attribute__((ext_vector_type(16))) __bf16        v16bf;
typedef __attribute__((ext_vector_type(8)))  float         v8f;
typedef __attribute__((ext_vector_type(8)))  unsigned int  v8u;

#define NBATCH 16
#define MELS   228
#define MELPAD 256    // 16 tiles of 16 mel rows (pad rows have zero weights)
#define BN     64     // spec rows (GEMM N columns) per workgroup
#define KC     32     // K chunk per stage = WMMA K depth
#define LSTR   36     // LDS row stride in bf16 (72 B = 18 banks)

// ---------------- reflect-pad + Hann window + radix-2 FFT + |X|^2 -> bf16 ---
__global__ void fft_power_kernel(const float* __restrict__ x, bf16_t* __restrict__ spec,
                                 int T, int n, int logn, int hop, int pad,
                                 int n_frames, int Kpad)
{
    extern __shared__ float2 lds[];
    float2* data = lds;           // [n]
    float2* tw   = lds + n;       // [n/2]
    const int tid    = threadIdx.x;
    const int frame  = blockIdx.x % n_frames;
    const int b      = blockIdx.x / n_frames;
    const float* xb  = x + (size_t)b * T;
    const int half_n = n >> 1;

    for (int k = tid; k < half_n; k += blockDim.x) {
        float ang = -6.283185307179586f * (float)k / (float)n;
        float s, c;
        __sincosf(ang, &s, &c);
        tw[k] = make_float2(c, s);
    }
    const float wscale = 6.283185307179586f / (float)n;
    for (int j = tid; j < n; j += blockDim.x) {
        int p = frame * hop + j - pad;
        if (p < 0)  p = -p;
        if (p >= T) p = 2 * T - 2 - p;
        float w = 0.5f - 0.5f * __cosf(wscale * (float)j);
        int r = (int)(__brev((unsigned)j) >> (32 - logn));
        data[r] = make_float2(xb[p] * w, 0.0f);
    }
    __syncthreads();

    for (int s = 1; s <= logn; ++s) {
        const int half  = 1 << (s - 1);
        const int tstep = n >> s;
        for (int j = tid; j < half_n; j += blockDim.x) {
            int pos = j & (half - 1);
            int i1  = ((j >> (s - 1)) << s) + pos;
            int i2  = i1 + half;
            float2 t = tw[pos * tstep];
            float2 u = data[i1];
            float2 v = data[i2];
            float2 tv = make_float2(t.x * v.x - t.y * v.y, t.x * v.y + t.y * v.x);
            data[i1] = make_float2(u.x + tv.x, u.y + tv.y);
            data[i2] = make_float2(u.x - tv.x, u.y - tv.y);
        }
        __syncthreads();
    }

    bf16_t* out = spec + (size_t)blockIdx.x * Kpad;
    const int F = half_n + 1;
    for (int f = tid; f < Kpad; f += blockDim.x) {
        float pwr = 0.0f;
        if (f < F) {
            float2 v = data[f];
            pwr = v.x * v.x + v.y * v.y;
        }
        out[f] = (bf16_t)pwr;
    }
}

// ---------------- slaney mel filterbank W[MELPAD][Kpad] (bf16, row-major) ---
__device__ __forceinline__ float mel_to_hz_dev(float m) {
    const float f_sp    = 200.0f / 3.0f;
    const float logstep = 0.06875177742094912f; // ln(6.4)/27
    return (m < 15.0f) ? f_sp * m : 1000.0f * expf(logstep * (m - 15.0f));
}

__global__ void fill_w_kernel(bf16_t* __restrict__ W, int n, int Kpad)
{
    const size_t idx   = (size_t)blockIdx.x * blockDim.x + threadIdx.x;
    const size_t total = (size_t)MELPAD * Kpad;
    if (idx >= total) return;
    const int m = (int)(idx / Kpad);
    const int k = (int)(idx % Kpad);
    const int F = n / 2 + 1;
    float wgt = 0.0f;
    if (m < MELS && k < F) {
        const float logstep = 0.06875177742094912f;           // ln(6.4)/27
        const float mel_max = 15.0f + logf(22.05f) / logstep; // hz_to_mel(22050)
        const float step    = mel_max / 229.0f;               // linspace(0, mel_max, 230)
        float f0 = mel_to_hz_dev(step * (float)m);
        float f1 = mel_to_hz_dev(step * (float)(m + 1));
        float f2 = mel_to_hz_dev(step * (float)(m + 2));
        float fr = (float)k * (44100.0f / (float)n);
        float lower = (fr - f0) / (f1 - f0);
        float upper = (f2 - fr) / (f2 - f1);
        float tri   = fmaxf(0.0f, fminf(lower, upper));
        wgt = tri * 2.0f / (f2 - f0);                         // slaney area norm
    }
    W[(size_t)m * Kpad + k] = (bf16_t)wgt;
}

// ---------------- mel GEMM: D[mel][row] = sum_k W[mel][k] * spec[row][k] ----
// A = W (16x32 fragments, M = mel), B = spec^T (32x16 fragments, N = frame row).
// Both fragments consume K-contiguous bf16 pairs -> 4B LDS dword reads.
// One 256-thread block = 8 waves computes a 256 x 64 output slab:
//   wave w -> N tile (w & 3), mel tiles (w>>2)*8 .. (w>>2)*8+7
// => 8 v8f accumulators per wave (64 VGPRs), no spills.
__global__ void __launch_bounds__(256)
mel_gemm_bf16_wmma(const bf16_t* __restrict__ W,
                   const bf16_t* __restrict__ spec,
                   float* __restrict__ melT, int rows, int Kpad)
{
    extern __shared__ char smem[];
    bf16_t* Ws = (bf16_t*)smem;                                  // [MELPAD][LSTR]
    bf16_t* Bs = (bf16_t*)(smem + (size_t)MELPAD * LSTR * 2);    // [BN][LSTR]

    const int tid   = threadIdx.x;
    const int wv    = tid >> 5;
    const int nt    = wv & 3;            // N tile within block
    const int mh    = wv >> 2;           // mel half: tiles mh*8 .. mh*8+7
    const int lane  = tid & 31;
    const int h     = lane >> 4;         // lane half
    const int ln    = lane & 15;
    const int nbase = blockIdx.x * BN;

    v8f acc[8];
#pragma unroll
    for (int mt = 0; mt < 8; ++mt)
        acc[mt] = (v8f){0.f, 0.f, 0.f, 0.f, 0.f, 0.f, 0.f, 0.f};

    for (int k0 = 0; k0 < Kpad; k0 += KC) {
        // ---- stage W chunk: 256 x 32 bf16, coalesced 8B loads ----
        for (int i = tid; i < MELPAD * (KC / 4); i += 256) {
            int m = i >> 3;              // 8 uint2 chunks per row
            int o = (i & 7) * 4;
            uint2 val = *(const uint2*)(W + (size_t)m * Kpad + k0 + o);
            *(uint2*)(Ws + m * LSTR + o) = val;
        }
        // ---- stage spec chunk: 64 rows x 32 bf16, zero-pad OOB rows ----
        for (int i = tid; i < BN * (KC / 4); i += 256) {
            int rr = i >> 3;
            int o  = (i & 7) * 4;
            int grow = nbase + rr;
            uint2 val = make_uint2(0u, 0u);
            if (grow < rows) val = *(const uint2*)(spec + (size_t)grow * Kpad + k0 + o);
            *(uint2*)(Bs + rr * LSTR + o) = val;
        }
        // prefetch next K chunk while this one is consumed
        if (k0 + KC < Kpad) {
            __builtin_prefetch(W + (size_t)(tid < MELPAD ? tid : 0) * Kpad + k0 + KC, 0, 0);
            int pr = nbase + (tid >> 2);
            if (pr < rows)
                __builtin_prefetch(spec + (size_t)pr * Kpad + k0 + KC + (tid & 3) * 8, 0, 0);
        }
        __syncthreads();

        // ---- B fragment (spec) built once per wave, reused for 8 tiles ----
        v8u bu;
#pragma unroll
        for (int v = 0; v < 8; ++v) {
            int kb = ((v >> 2) << 4) + (h << 3) + ((v & 3) << 1);
            bu[v] = *(const unsigned int*)(Bs + (nt * 16 + ln) * LSTR + kb);
        }
        v16bf bfrag = __builtin_bit_cast(v16bf, bu);

#pragma unroll
        for (int mt = 0; mt < 8; ++mt) {
            v8u au;
#pragma unroll
            for (int v = 0; v < 8; ++v) {
                int kb = ((v >> 2) << 4) + (h << 3) + ((v & 3) << 1);
                au[v] = *(const unsigned int*)(Ws + ((mh * 8 + mt) * 16 + ln) * LSTR + kb);
            }
            v16bf afrag = __builtin_bit_cast(v16bf, au);
            acc[mt] = __builtin_amdgcn_wmma_f32_16x16x32_bf16(
                          /*neg_a=*/false, afrag, /*neg_b=*/false, bfrag,
                          /*c_mod=*/(short)0, acc[mt],
                          /*reuse_a=*/false, /*reuse_b=*/false);
        }
        __syncthreads();
    }

    // C layout: VGPR r -> M = r + 8*h (mel), N = ln (frame row)
    const int nn = nbase + nt * 16 + ln;
    if (nn < rows) {
#pragma unroll
        for (int mt = 0; mt < 8; ++mt) {
            float* op = melT + (size_t)((mh * 8 + mt) * 16 + h * 8) * rows + nn;
#pragma unroll
            for (int r2 = 0; r2 < 8; ++r2)
                op[(size_t)r2 * rows] = acc[mt][r2];
        }
    }
}

// ---------------- sum |log(max(a,eps)) - log(max(b,eps))| -------------------
// melT layout: [MELPAD][rows], only m < MELS contributes.
__global__ void diff_reduce_kernel(const float* __restrict__ A, const float* __restrict__ B,
                                   int rows, float* __restrict__ acc)
{
    __shared__ float red[256];
    const size_t total  = (size_t)MELS * rows;
    const size_t stride = (size_t)gridDim.x * blockDim.x;
    float s = 0.0f;
    for (size_t i = (size_t)blockIdx.x * blockDim.x + threadIdx.x; i < total; i += stride) {
        float pa = A[i];
        float pb = B[i];
        s += fabsf(logf(fmaxf(pa, 1e-9f)) - logf(fmaxf(pb, 1e-9f)));
    }
    red[threadIdx.x] = s;
    __syncthreads();
    for (int off = 128; off > 0; off >>= 1) {
        if ((int)threadIdx.x < off) red[threadIdx.x] += red[threadIdx.x + off];
        __syncthreads();
    }
    if (threadIdx.x == 0) atomicAdd(acc, red[0]);
}

__global__ void zero3_kernel(float* acc) {
    if (threadIdx.x < 3) acc[threadIdx.x] = 0.0f;
}

__global__ void finalize_kernel(const float* __restrict__ acc, float* __restrict__ out,
                                float i0, float i1, float i2)
{
    out[0] = (acc[0] * i0 + acc[1] * i1 + acc[2] * i2) * (1.0f / 3.0f);
}

// ---------------------------------------------------------------------------
extern "C" void kernel_launch(void* const* d_in, const int* in_sizes, int n_in,
                              void* d_out, int out_size, void* d_ws, size_t ws_size,
                              hipStream_t stream)
{
    const float* pred = (const float*)d_in[0];
    const float* targ = (const float*)d_in[1];
    float* out = (float*)d_out;
    const int T = in_sizes[0] / NBATCH;   // 352800

    // workspace layout (peak ~74 MB)
    char*   ws   = (char*)d_ws;
    float*  acc  = (float*)ws;                          // 3 floats
    bf16_t* W    = (bf16_t*)(ws + 256);                 // <= 256*2080*2  ~ 1.1 MB
    bf16_t* spec = (bf16_t*)(ws + ((size_t)2  << 20));  // <= 22064*544*2 ~ 24.0 MB
    float*  melA = (float*)(ws + ((size_t)27 << 20));   // 256*22064*4   ~ 22.6 MB
    float*  melB = (float*)(ws + ((size_t)51 << 20));   // 256*22064*4   ~ 22.6 MB

    zero3_kernel<<<1, 32, 0, stream>>>(acc);

    const int NS[3]   = {1024, 2048, 4096};
    const int LOGN[3] = {10, 11, 12};
    float invCnt[3];

    const size_t gemm_lds = (size_t)(MELPAD + BN) * LSTR * 2;   // 23040 B

    for (int r = 0; r < 3; ++r) {
        const int n    = NS[r];
        const int logn = LOGN[r];
        const int hop  = n / 4;
        const int pad  = n / 2;
        const int nf   = 1 + T / hop;
        const int rows = NBATCH * nf;             // multiple of 16 (B=16)
        const int F    = n / 2 + 1;
        const int Kpad = ((F + 31) / 32) * 32;
        invCnt[r] = 1.0f / ((float)rows * (float)MELS);

        const size_t fft_lds = (size_t)(n + n / 2) * sizeof(float2);  // <= 48 KB

        fill_w_kernel<<<(unsigned)(((size_t)MELPAD * Kpad + 255) / 256), 256, 0, stream>>>(W, n, Kpad);

        const int gblks = (rows + BN - 1) / BN;

        fft_power_kernel<<<rows, 256, fft_lds, stream>>>(pred, spec, T, n, logn, hop, pad, nf, Kpad);
        mel_gemm_bf16_wmma<<<gblks, 256, gemm_lds, stream>>>(W, spec, melA, rows, Kpad);

        fft_power_kernel<<<rows, 256, fft_lds, stream>>>(targ, spec, T, n, logn, hop, pad, nf, Kpad);
        mel_gemm_bf16_wmma<<<gblks, 256, gemm_lds, stream>>>(W, spec, melB, rows, Kpad);

        diff_reduce_kernel<<<1024, 256, 0, stream>>>(melA, melB, rows, acc + r);
    }

    finalize_kernel<<<1, 1, 0, stream>>>(acc, out, invCnt[0], invCnt[1], invCnt[2]);
}
